// ChebNetwork_75522704933515
// MI455X (gfx1250) — matrix operand order repaired and checked by reading, not compiled, not run on hardware
//
#include <hip/hip_runtime.h>
#include <cstdint>
#include <cstddef>

typedef _Float16 half_t;
typedef half_t v16h __attribute__((ext_vector_type(16)));
typedef float  v8f  __attribute__((ext_vector_type(8)));

// ---------------------------------------------------------------------------
// Degree accumulation: deg[src[e]] += w[e]
// ---------------------------------------------------------------------------
__global__ __launch_bounds__(256) void deg_kernel(const long long* __restrict__ src,
                                                  const float* __restrict__ ew,
                                                  float* __restrict__ deg, int E) {
    int e = blockIdx.x * 256 + threadIdx.x;
    if (e >= E) return;
    atomicAdd(&deg[(int)src[e]], ew[e]);
}

// ---------------------------------------------------------------------------
// norm_w[e] = -dinv[src] * w * dinv[tgt],  dinv = deg>0 ? rsqrt(max(deg,1e-12)) : 0
// ---------------------------------------------------------------------------
__global__ __launch_bounds__(256) void normw_kernel(const long long* __restrict__ src,
                                                    const long long* __restrict__ tgt,
                                                    const float* __restrict__ ew,
                                                    const float* __restrict__ deg,
                                                    float* __restrict__ nw, int E) {
    int e = blockIdx.x * 256 + threadIdx.x;
    if (e >= E) return;
    float ds = deg[(int)src[e]];
    float dt = deg[(int)tgt[e]];
    float is = ds > 0.0f ? __frsqrt_rn(fmaxf(ds, 1e-12f)) : 0.0f;
    float it = dt > 0.0f ? __frsqrt_rn(fmaxf(dt, 1e-12f)) : 0.0f;
    nw[e] = -is * ew[e] * it;
}

// ---------------------------------------------------------------------------
// Weight prep: W [2][FIN][64] f32  ->  Wt [2][64][FIN] f16 (transposed, n-major)
// so each WMMA B-fragment is one contiguous 32B load per lane.
// ---------------------------------------------------------------------------
__global__ __launch_bounds__(256) void prep_w_kernel(const float* __restrict__ W,
                                                     half_t* __restrict__ Wt, int FIN) {
    int tid = blockIdx.x * 256 + threadIdx.x;
    int total = 2 * FIN * 64;
    if (tid >= total) return;
    int t   = tid / (FIN * 64);
    int rem = tid - t * FIN * 64;
    int k   = rem >> 6;   // row of W (K index)
    int n   = rem & 63;   // col of W (N index)
    Wt[(size_t)(t * 64 + n) * FIN + k] = (half_t)W[tid];
}

// ---------------------------------------------------------------------------
// SpMM: tx1[tgt[e], :] += nw[e] * x[src[e], :]
// F/4 lanes per edge; each lane: one float4 gather + 4 f32 atomics.
// F=128 -> 1 wave/edge, F=64 -> 2 edges/wave.
// ---------------------------------------------------------------------------
template <int F>
__global__ __launch_bounds__(256) void spmm_kernel(const float* __restrict__ x,
                                                   const long long* __restrict__ src,
                                                   const long long* __restrict__ tgt,
                                                   const float* __restrict__ nw,
                                                   float* __restrict__ tx1, int E) {
    constexpr int LPE = F / 4;        // lanes per edge
    constexpr int EPB = 256 / LPE;    // edges per block
    int sub  = threadIdx.x / LPE;
    int fl   = threadIdx.x % LPE;     // feature-lane within edge
    int edge = blockIdx.x * EPB + sub;
    if (edge >= E) return;
    long long s = src[edge];
    long long t = tgt[edge];
    float w = nw[edge];
    const float* xp = x   + (size_t)s * F + fl * 4;
    float*       op = tx1 + (size_t)t * F + fl * 4;
    float4 v = *(const float4*)xp;
    atomicAdd(op + 0, w * v.x);
    atomicAdd(op + 1, w * v.y);
    atomicAdd(op + 2, w * v.z);
    atomicAdd(op + 3, w * v.w);
}

// ---------------------------------------------------------------------------
// Fused dual-GEMM + bias + sigmoid:
//   out = sigmoid(X0 @ W[0] + X1 @ W[1] + b), out is [N,64]
// One wave owns one 16-row M-tile across all 64 cols: per K-step it converts
// the A fragment ONCE and issues 4 independent WMMAs (one per 16-col tile).
// ---------------------------------------------------------------------------
template <int FIN>
__global__ __launch_bounds__(256) void gemm_fused_kernel(const float* __restrict__ X0,
                                                         const float* __restrict__ X1,
                                                         const half_t* __restrict__ Wt, // [2][64][FIN]
                                                         const float* __restrict__ bias,
                                                         float* __restrict__ out, int N) {
    const int lane  = threadIdx.x & 31;
    const int wave  = threadIdx.x >> 5;
    const int mtile = blockIdx.x * 8 + wave;
    const int rowBase = mtile * 16;
    if (rowBase >= N) return;                  // wave-uniform exit (no WMMA executed)

    const int nlo = lane & 15;                 // column-within-tile / row-within-tile
    const int hi  = lane >> 4;                 // K-split lane group
    const int m   = rowBase + nlo;             // A-matrix row for this lane

    v8f acc0 = {}, acc1 = {}, acc2 = {}, acc3 = {};

    #pragma unroll
    for (int t = 0; t < 2; ++t) {
        const float*  X  = t ? X1 : X0;
        const half_t* Wb = Wt + (size_t)t * 64 * FIN;
        #pragma unroll
        for (int ks = 0; ks < FIN; ks += 32) {
            // A fragment: half j <- K = ks + 16*hi + j
            const float* ap = X + (size_t)m * FIN + ks + hi * 16;
            float4 a0 = *(const float4*)(ap + 0);
            float4 a1 = *(const float4*)(ap + 4);
            float4 a2 = *(const float4*)(ap + 8);
            float4 a3 = *(const float4*)(ap + 12);
            v16h a;
            a[0]  = (half_t)a0.x; a[1]  = (half_t)a0.y; a[2]  = (half_t)a0.z; a[3]  = (half_t)a0.w;
            a[4]  = (half_t)a1.x; a[5]  = (half_t)a1.y; a[6]  = (half_t)a1.z; a[7]  = (half_t)a1.w;
            a[8]  = (half_t)a2.x; a[9]  = (half_t)a2.y; a[10] = (half_t)a2.z; a[11] = (half_t)a2.w;
            a[12] = (half_t)a3.x; a[13] = (half_t)a3.y; a[14] = (half_t)a3.z; a[15] = (half_t)a3.w;

            // B fragments for the 4 column tiles: half j <- K = ks + 16*hi + j
            const half_t* wp = Wb + ks + hi * 16;
            v16h b0 = *(const v16h*)(wp + (size_t)(0 * 16 + nlo) * FIN);
            v16h b1 = *(const v16h*)(wp + (size_t)(1 * 16 + nlo) * FIN);
            v16h b2 = *(const v16h*)(wp + (size_t)(2 * 16 + nlo) * FIN);
            v16h b3 = *(const v16h*)(wp + (size_t)(3 * 16 + nlo) * FIN);

            acc0 = __builtin_amdgcn_wmma_f32_16x16x32_f16(false, a, false, b0, (short)0, acc0, false, false);
            acc1 = __builtin_amdgcn_wmma_f32_16x16x32_f16(false, a, false, b1, (short)0, acc1, false, false);
            acc2 = __builtin_amdgcn_wmma_f32_16x16x32_f16(false, a, false, b2, (short)0, acc2, false, false);
            acc3 = __builtin_amdgcn_wmma_f32_16x16x32_f16(false, a, false, b3, (short)0, acc3, false, false);
        }
    }

    // Epilogue: C/D layout -> VGPR r of tile nt holds row (rowBase + r + 8*hi),
    // col nt*16 + nlo.
    float bn0 = bias[0 * 16 + nlo];
    float bn1 = bias[1 * 16 + nlo];
    float bn2 = bias[2 * 16 + nlo];
    float bn3 = bias[3 * 16 + nlo];
    #pragma unroll
    for (int r = 0; r < 8; ++r) {
        size_t rowOff = (size_t)(rowBase + r + hi * 8) * 64;
        out[rowOff + 0 * 16 + nlo] = 1.0f / (1.0f + __expf(-(acc0[r] + bn0)));
        out[rowOff + 1 * 16 + nlo] = 1.0f / (1.0f + __expf(-(acc1[r] + bn1)));
        out[rowOff + 2 * 16 + nlo] = 1.0f / (1.0f + __expf(-(acc2[r] + bn2)));
        out[rowOff + 3 * 16 + nlo] = 1.0f / (1.0f + __expf(-(acc3[r] + bn3)));
    }
}

// ---------------------------------------------------------------------------
// Host orchestration
// ---------------------------------------------------------------------------
static inline size_t alignup256(size_t v) { return (v + 255) & ~(size_t)255; }

extern "C" void kernel_launch(void* const* d_in, const int* in_sizes, int n_in,
                              void* d_out, int out_size, void* d_ws, size_t ws_size,
                              hipStream_t stream) {
    (void)n_in; (void)out_size; (void)ws_size;

    const float*     x   = (const float*)d_in[0];
    const long long* ei  = (const long long*)d_in[1];
    const float*     ew  = (const float*)d_in[2];
    const float*     W1  = (const float*)d_in[3];
    const float*     b1  = (const float*)d_in[4];
    const float*     W2  = (const float*)d_in[5];
    const float*     b2  = (const float*)d_in[6];
    const float*     W3  = (const float*)d_in[7];
    const float*     b3  = (const float*)d_in[8];
    const float*     W4  = (const float*)d_in[9];
    const float*     b4  = (const float*)d_in[10];

    const int N = in_sizes[0] / 128;     // 100000
    const int E = in_sizes[2];           // 3200000
    const long long* src = ei;           // edge_index[0] (== "row")
    const long long* tgt = ei + E;       // edge_index[1] (== "col")

    // Workspace carve-out
    char* ws = (char*)d_ws;
    size_t off = 0;
    float* deg = (float*)(ws + off);  off += alignup256((size_t)N * 4);
    float* nw  = (float*)(ws + off);  off += alignup256((size_t)E * 4);
    float* tx1 = (float*)(ws + off);  off += alignup256((size_t)N * 128 * 4);
    float* hA  = (float*)(ws + off);  off += alignup256((size_t)N * 64 * 4);
    float* hB  = (float*)(ws + off);  off += alignup256((size_t)N * 64 * 4);
    half_t* Wt1 = (half_t*)(ws + off); off += alignup256((size_t)2 * 64 * 128 * 2);
    half_t* Wt2 = (half_t*)(ws + off); off += alignup256((size_t)2 * 64 * 64 * 2);
    half_t* Wt3 = (half_t*)(ws + off); off += alignup256((size_t)2 * 64 * 64 * 2);
    half_t* Wt4 = (half_t*)(ws + off); off += alignup256((size_t)2 * 64 * 64 * 2);

    const int eb     = (E + 255) / 256;          // 1-thread-per-edge grids
    const int swb128 = (E + 7) / 8;              // spmm<128>: 8 edges/block
    const int swb64  = (E + 15) / 16;            // spmm<64>: 16 edges/block
    const int mtiles = (N + 15) / 16;            // 6250
    const int gb     = (mtiles + 7) / 8;         // one wave per 16-row M-tile

    // --- normalization weights (shared by all 4 layers) ---
    hipMemsetAsync(deg, 0, (size_t)N * 4, stream);
    deg_kernel<<<eb, 256, 0, stream>>>(src, ew, deg, E);
    normw_kernel<<<eb, 256, 0, stream>>>(src, tgt, ew, deg, nw, E);

    // --- weight conversion/transposition to f16 ---
    prep_w_kernel<<<(2 * 128 * 64 + 255) / 256, 256, 0, stream>>>(W1, Wt1, 128);
    prep_w_kernel<<<(2 * 64 * 64 + 255) / 256, 256, 0, stream>>>(W2, Wt2, 64);
    prep_w_kernel<<<(2 * 64 * 64 + 255) / 256, 256, 0, stream>>>(W3, Wt3, 64);
    prep_w_kernel<<<(2 * 64 * 64 + 255) / 256, 256, 0, stream>>>(W4, Wt4, 64);

    // --- layer 1: Fin=128 ---
    hipMemsetAsync(tx1, 0, (size_t)N * 128 * 4, stream);
    spmm_kernel<128><<<swb128, 256, 0, stream>>>(x, src, tgt, nw, tx1, E);
    gemm_fused_kernel<128><<<gb, 256, 0, stream>>>(x, tx1, Wt1, b1, hA, N);

    // --- layer 2: Fin=64 ---
    hipMemsetAsync(tx1, 0, (size_t)N * 64 * 4, stream);
    spmm_kernel<64><<<swb64, 256, 0, stream>>>(hA, src, tgt, nw, tx1, E);
    gemm_fused_kernel<64><<<gb, 256, 0, stream>>>(hA, tx1, Wt2, b2, hB, N);

    // --- layer 3 ---
    hipMemsetAsync(tx1, 0, (size_t)N * 64 * 4, stream);
    spmm_kernel<64><<<swb64, 256, 0, stream>>>(hB, src, tgt, nw, tx1, E);
    gemm_fused_kernel<64><<<gb, 256, 0, stream>>>(hB, tx1, Wt3, b3, hA, N);

    // --- layer 4 -> d_out ---
    hipMemsetAsync(tx1, 0, (size_t)N * 64 * 4, stream);
    spmm_kernel<64><<<swb64, 256, 0, stream>>>(hA, src, tgt, nw, tx1, E);
    gemm_fused_kernel<64><<<gb, 256, 0, stream>>>(hA, tx1, Wt4, b4, (float*)d_out, N);
}
